// GatedAdapters_18081812316349
// MI455X (gfx1250) — compile-verified, matrix operand order checked by reading, not run on hardware
//
#include <hip/hip_runtime.h>
#include <stdint.h>

typedef __attribute__((ext_vector_type(16))) __bf16 v16bf;
typedef __attribute__((ext_vector_type(8)))  float  v8f;

static constexpr int S_   = 2048;
static constexpr int B_   = 8;
static constexpr int D_   = 1024;
static constexpr int E_   = 8;
static constexpr int H_   = 512;
static constexpr int MTOT = S_ * B_;      // 16384 tokens
static constexpr int N1   = E_ * H_;      // 4096 (stacked fc1 outputs)
static constexpr int SCH  = 16;           // max-pool S chunks
static constexpr int CHUNK = 4096;        // M chunk for GEMM passes

struct __attribute__((aligned(16))) U4 { uint32_t x, y, z, w; };
union FragBF { U4 q[2]; v16bf v; };
union BStage { U4 q[2]; unsigned short s[16]; };

__device__ __forceinline__ unsigned short f2bf(float f) {
  union { float f; uint32_t u; } c; c.f = f;
  uint32_t u = c.u;
  uint32_t r = u + 0x7FFFu + ((u >> 16) & 1u);   // round-to-nearest-even
  return (unsigned short)(r >> 16);
}

// ---------------- max-pool over sequence (2-stage) ----------------
__global__ void maxpool_part(const float* __restrict__ x,
                             const unsigned char* __restrict__ pad,
                             float* __restrict__ part) {
  int id = blockIdx.x * blockDim.x + threadIdx.x;
  if (id >= SCH * B_ * D_) return;
  int ch  = id / (B_ * D_);
  int rem = id % (B_ * D_);
  int b   = rem / D_;
  int s0  = ch * (S_ / SCH);
  float m = -3.0e38f;
  for (int s = s0; s < s0 + S_ / SCH; ++s) {
    float v = x[(size_t)s * (B_ * D_) + rem];
    if (pad[b * S_ + s]) v -= 1e30f;
    m = fmaxf(m, v);
  }
  part[id] = m;
}

__global__ void maxpool_fin(const float* __restrict__ part, float* __restrict__ sent) {
  int id = blockIdx.x * blockDim.x + threadIdx.x;
  if (id >= B_ * D_) return;
  float m = part[id];
  for (int c = 1; c < SCH; ++c) m = fmaxf(m, part[c * B_ * D_ + id]);
  sent[id] = m;
}

// ---------------- gate logits + softmax + renorm ----------------
__global__ void gates_kernel(const float* __restrict__ sent,
                             const float* __restrict__ gw,
                             float* __restrict__ logits_out,
                             float* __restrict__ gates) {
  __shared__ float sl[B_ * E_];
  int t = threadIdx.x;                 // 64 threads: t = b*8 + e
  int b = t >> 3, e = t & 7;
  float acc = 0.f;
  for (int d = 0; d < D_; ++d) acc += sent[b * D_ + d] * gw[e * D_ + d];
  logits_out[t] = acc;
  sl[t] = acc;
  __syncthreads();
  float mx = -3e38f;
  for (int j = 0; j < E_; ++j) mx = fmaxf(mx, sl[b * E_ + j]);
  float se = 0.f;
  for (int j = 0; j < E_; ++j) se += __expf(sl[b * E_ + j] - mx);
  float g = __expf(acc - mx) / se;     // softmax
  gates[t] = g / (1.0f + 1e-9f);       // renorm (softmax row-sum == 1)
}

// ---------------- gated fc2 bias:  bias2g[b,d] = sum_e gates[b,e]*b2[e,d] ----------------
__global__ void bias2_kernel(const float* __restrict__ gates,
                             const float* __restrict__ b2,
                             float* __restrict__ bias2g) {
  int id = blockIdx.x * blockDim.x + threadIdx.x;
  if (id >= B_ * D_) return;
  int b = id / D_, d = id % D_;
  float a = 0.f;
  for (int e = 0; e < E_; ++e) a += gates[b * E_ + e] * b2[e * D_ + d];
  bias2g[id] = a;
}

// ---------------- LayerNorm per token, output bf16 ----------------
__global__ __launch_bounds__(256)
void ln_kernel(const float* __restrict__ x, const float* __restrict__ gamma,
               const float* __restrict__ beta, unsigned short* __restrict__ xn) {
  int row = blockIdx.x;
  int t = threadIdx.x;
  const float* xr = x + (size_t)row * D_;
  float4 v = ((const float4*)xr)[t];           // 4 elems / thread, 256 threads
  float s  = v.x + v.y + v.z + v.w;
  float s2 = v.x * v.x + v.y * v.y + v.z * v.z + v.w * v.w;
  for (int off = 16; off > 0; off >>= 1) {
    s  += __shfl_xor(s,  off, 32);
    s2 += __shfl_xor(s2, off, 32);
  }
  __shared__ float a1[8], a2[8];
  __shared__ float mu_s, rs_s;
  int wave = t >> 5, lane = t & 31;
  if (lane == 0) { a1[wave] = s; a2[wave] = s2; }
  __syncthreads();
  if (t == 0) {
    float S1 = 0.f, S2 = 0.f;
    for (int i = 0; i < 8; ++i) { S1 += a1[i]; S2 += a2[i]; }
    float mu = S1 * (1.0f / D_);
    float var = S2 * (1.0f / D_) - mu * mu;
    mu_s = mu; rs_s = rsqrtf(var + 1e-5f);
  }
  __syncthreads();
  float mu = mu_s, rs = rs_s;
  int c0 = t * 4;
  unsigned short o0 = f2bf((v.x - mu) * rs * gamma[c0 + 0] + beta[c0 + 0]);
  unsigned short o1 = f2bf((v.y - mu) * rs * gamma[c0 + 1] + beta[c0 + 1]);
  unsigned short o2 = f2bf((v.z - mu) * rs * gamma[c0 + 2] + beta[c0 + 2]);
  unsigned short o3 = f2bf((v.w - mu) * rs * gamma[c0 + 3] + beta[c0 + 3]);
  uint32_t lo = (uint32_t)o0 | ((uint32_t)o1 << 16);
  uint32_t hi = (uint32_t)o2 | ((uint32_t)o3 << 16);
  uint2 pk; pk.x = lo; pk.y = hi;
  ((uint2*)(xn + (size_t)row * D_))[t] = pk;
}

// ---------------- weight conversions to bf16 ----------------
// W1cat[k][n] = w1[e=n>>9][k][h=n&511], row-major [1024][4096]
__global__ void convw1_kernel(const float* __restrict__ w1, unsigned short* __restrict__ w1b) {
  size_t id = (size_t)blockIdx.x * blockDim.x + threadIdx.x;
  if (id >= (size_t)D_ * N1) return;
  int k = (int)(id / N1), n = (int)(id % N1);
  int e = n >> 9, h = n & (H_ - 1);
  w1b[id] = f2bf(w1[((size_t)e * D_ + k) * H_ + h]);
}
// W2cat[k=e*512+h][d] = w2[e][h][d]  (already contiguous) -> plain convert
__global__ void convw2_kernel(const float* __restrict__ w2, unsigned short* __restrict__ w2b) {
  size_t id = (size_t)blockIdx.x * blockDim.x + threadIdx.x;
  if (id >= (size_t)N1 * D_) return;
  w2b[id] = f2bf(w2[id]);
}

// issue one 32-byte async global->LDS copy (two b128 ops; offset applies to both sides)
__device__ __forceinline__ void async_copy32(uint32_t lds_off, const unsigned short* src) {
  uint64_t sa = (uint64_t)src;
  asm volatile("global_load_async_to_lds_b128 %0, %1, off"
               :: "v"(lds_off), "v"(sa) : "memory");
  asm volatile("global_load_async_to_lds_b128 %0, %1, off offset:16"
               :: "v"(lds_off), "v"(sa) : "memory");
}
__device__ __forceinline__ void wait_async0() {
  asm volatile("s_wait_asynccnt 0x0" ::: "memory");
}

// ---------------- bf16 WMMA GEMM, 128x128 tile, BK=32, double-buffered LDS ----------------
// MODE 0 (fc1):  C = A[M,K] * B[K,N] ; epilogue: +b1, relu, *gates -> bf16 Hout
// MODE 1 (fc2):  C = A[M,K] * B[K,N] ; epilogue: +residual +gated-bias -> f32 out
template <int MODE>
__global__ __launch_bounds__(256)
void gemm_kernel(const unsigned short* __restrict__ A,
                 const unsigned short* __restrict__ B,
                 int K, int N, int row0,
                 const float* __restrict__ gates,
                 const float* __restrict__ b1,
                 unsigned short* __restrict__ Hout,
                 const float* __restrict__ xres,
                 const float* __restrict__ bias2g,
                 float* __restrict__ out) {
  constexpr int BK  = 32;
  constexpr int LDR = 40;   // LDS row pitch in bf16 elems (80B, 16B-aligned)
  __shared__ alignas(16) unsigned short lsA[2][128 * LDR];
  __shared__ alignas(16) unsigned short lsB[2][128 * LDR];

  const int t    = threadIdx.x;
  const int lane = t & 31;
  const int wave = t >> 5;
  const int kh   = (lane >> 4) & 1;     // lane half selects K sub-pattern
  const int l16  = lane & 15;
  const int wm   = wave >> 2;           // 2 wave rows  (64 each)
  const int wn   = wave & 3;            // 4 wave cols  (32 each)
  const int rowBase = blockIdx.y * 128;
  const int colBase = blockIdx.x * 128;

  // global staging: A tile 128x32 (2 threads/row, 16 bf16 each),
  //                 B tile 32x128 (8 threads/row, 16 bf16 each)
  const int arow = t >> 1, acol = (t & 1) * 16;
  const int brow = t >> 3, bcol = (t & 7) * 16;
  const unsigned short* gA = A + (size_t)(rowBase + arow) * K + acol;
  const unsigned short* gB = B + (size_t)brow * N + colBase + bcol;

  // LDS byte offsets for this thread's A destination in each buffer
  // (low 32 bits of the generic pointer == workgroup-relative LDS address)
  const uint32_t dA0 = (uint32_t)(uintptr_t)&lsA[0][arow * LDR + acol];
  const uint32_t dA1 = (uint32_t)(uintptr_t)&lsA[1][arow * LDR + acol];

  v8f zero = {};
  v8f acc[4][2];
#pragma unroll
  for (int i = 0; i < 4; ++i)
#pragma unroll
    for (int j = 0; j < 2; ++j) acc[i][j] = zero;

  const int nK = K / BK;

  // prologue: tile 0 -> lds buffer 0 (A async, B via regs w/ transpose)
  async_copy32(dA0, gA);
  {
    BStage rb;
    rb.q[0] = *(const U4*)gB;
    rb.q[1] = *(const U4*)(gB + 8);
#pragma unroll
    for (int j = 0; j < 16; ++j) lsB[0][(bcol + j) * LDR + brow] = rb.s[j];
  }
  wait_async0();
  __syncthreads();

  int cur = 0;
  for (int kt = 0; kt < nK; ++kt) {
    BStage rb;
    const bool more = (kt + 1 < nK);
    if (more) {
      // A tile kt+1: async straight into the other LDS buffer
      async_copy32(cur == 0 ? dA1 : dA0, gA + (size_t)(kt + 1) * BK);
      const unsigned short* pB = gB + (size_t)(kt + 1) * BK * N;
      rb.q[0] = *(const U4*)pB;
      rb.q[1] = *(const U4*)(pB + 8);
      if (kt + 2 < nK) {
        __builtin_prefetch(gB + (size_t)(kt + 2) * BK * N, 0, 0);   // global_prefetch_b8
      }
    }

    // fragment loads: per-lane row of 32 bf16, two 16B reads selected by lane-half
    FragBF af[4], bfr[2];
#pragma unroll
    for (int fm = 0; fm < 4; ++fm) {
      const uint32_t* rp = (const uint32_t*)&lsA[cur][(wm * 64 + fm * 16 + l16) * LDR];
      af[fm].q[0] = *(const U4*)(rp + 4 * kh);
      af[fm].q[1] = *(const U4*)(rp + 8 + 4 * kh);
    }
#pragma unroll
    for (int fn = 0; fn < 2; ++fn) {
      const uint32_t* rp = (const uint32_t*)&lsB[cur][(wn * 32 + fn * 16 + l16) * LDR];
      bfr[fn].q[0] = *(const U4*)(rp + 4 * kh);
      bfr[fn].q[1] = *(const U4*)(rp + 8 + 4 * kh);
    }
#pragma unroll
    for (int fm = 0; fm < 4; ++fm)
#pragma unroll
      for (int fn = 0; fn < 2; ++fn)
        acc[fm][fn] = __builtin_amdgcn_wmma_f32_16x16x32_bf16(
            false, af[fm].v, false, bfr[fn].v, (short)0, acc[fm][fn], false, false);

    if (more) {
      int nxt = cur ^ 1;
#pragma unroll
      for (int j = 0; j < 16; ++j) lsB[nxt][(bcol + j) * LDR + brow] = rb.s[j];
      wait_async0();        // A tile kt+1 resident in LDS
      __syncthreads();      // B stores drained via DScnt by barrier
      cur = nxt;
    }
  }

  // epilogue; C/D layout: lane l16 = col, VGPR v -> row 8*kh + v
#pragma unroll
  for (int fm = 0; fm < 4; ++fm) {
#pragma unroll
    for (int fn = 0; fn < 2; ++fn) {
#pragma unroll
      for (int v = 0; v < 8; ++v) {
        int r = rowBase + wm * 64 + fm * 16 + kh * 8 + v;   // chunk-local row
        int c = colBase + wn * 32 + fn * 16 + l16;
        float val = acc[fm][fn][v];
        if (MODE == 0) {
          val += b1[c];
          val = fmaxf(val, 0.0f);
          int e  = c >> 9;                  // expert = n / H
          int bb = (row0 + r) & (B_ - 1);   // token batch index
          val *= gates[bb * E_ + e];
          Hout[(size_t)r * N + c] = f2bf(val);
        } else {
          int tok = row0 + r;
          size_t idx = (size_t)tok * D_ + c;
          out[idx] = val + xres[idx] + bias2g[((tok & (B_ - 1)) << 10) + c];
        }
      }
    }
  }
}

extern "C" void kernel_launch(void* const* d_in, const int* in_sizes, int n_in,
                              void* d_out, int out_size, void* d_ws, size_t ws_size,
                              hipStream_t stream) {
  (void)in_sizes; (void)n_in; (void)out_size; (void)ws_size;
  const float*         x     = (const float*)d_in[0];
  const unsigned char* pad   = (const unsigned char*)d_in[1];
  const float*         gw    = (const float*)d_in[2];
  const float*         gamma = (const float*)d_in[3];
  const float*         beta  = (const float*)d_in[4];
  const float*         w1    = (const float*)d_in[5];
  const float*         b1    = (const float*)d_in[6];
  const float*         w2    = (const float*)d_in[7];
  const float*         b2    = (const float*)d_in[8];

  float* out        = (float*)d_out;
  float* logits_out = out + (size_t)MTOT * D_;   // gate_logits appended after out

  // workspace layout (256B-aligned slices)
  char* ws = (char*)d_ws;
  size_t off = 0;
  auto alloc = [&](size_t bytes) { size_t o = off; off += (bytes + 255) & ~(size_t)255; return o; };
  float*          sent   = (float*)(ws + alloc(sizeof(float) * B_ * D_));
  float*          gates  = (float*)(ws + alloc(sizeof(float) * B_ * E_));
  float*          bias2g = (float*)(ws + alloc(sizeof(float) * B_ * D_));
  float*          pmax   = (float*)(ws + alloc(sizeof(float) * SCH * B_ * D_));
  unsigned short* xnbf   = (unsigned short*)(ws + alloc(sizeof(short) * (size_t)MTOT * D_));
  unsigned short* w1b    = (unsigned short*)(ws + alloc(sizeof(short) * (size_t)D_ * N1));
  unsigned short* w2b    = (unsigned short*)(ws + alloc(sizeof(short) * (size_t)N1 * D_));
  unsigned short* Hb     = (unsigned short*)(ws + alloc(sizeof(short) * (size_t)CHUNK * N1));

  // gating path
  maxpool_part<<<(SCH * B_ * D_ + 255) / 256, 256, 0, stream>>>(x, pad, pmax);
  maxpool_fin<<<(B_ * D_ + 255) / 256, 256, 0, stream>>>(pmax, sent);
  gates_kernel<<<1, 64, 0, stream>>>(sent, gw, logits_out, gates);
  bias2_kernel<<<(B_ * D_ + 255) / 256, 256, 0, stream>>>(gates, b2, bias2g);

  // LN + bf16 conversions
  ln_kernel<<<MTOT, 256, 0, stream>>>(x, gamma, beta, xnbf);
  convw1_kernel<<<(int)(((size_t)D_ * N1 + 255) / 256), 256, 0, stream>>>(w1, w1b);
  convw2_kernel<<<(int)(((size_t)N1 * D_ + 255) / 256), 256, 0, stream>>>(w2, w2b);

  // chunked fc1 -> fc2
  for (int c0 = 0; c0 < MTOT; c0 += CHUNK) {
    dim3 g1(N1 / 128, CHUNK / 128);
    gemm_kernel<0><<<g1, 256, 0, stream>>>(xnbf + (size_t)c0 * D_, w1b, D_, N1, c0,
                                           gates, b1, Hb, nullptr, nullptr, nullptr);
    dim3 g2(D_ / 128, CHUNK / 128);
    gemm_kernel<1><<<g2, 256, 0, stream>>>(Hb, w2b, N1, D_, c0,
                                           gates, nullptr, nullptr, x, bias2g, out);
  }
}
